// SwinTransformerBlock_39135742001422
// MI455X (gfx1250) — matrix operand order repaired and checked by reading, not compiled, not run on hardware
//
#include <hip/hip_runtime.h>
#include <hip/hip_bf16.h>
#include <math.h>

// Problem constants (match reference)
#define B_    32
#define L_    4096
#define C_    256
#define NH_   8
#define HD_   32
#define HID_  1024
#define SHIFT_ 4

typedef __attribute__((ext_vector_type(16))) __bf16        v16bf;
typedef __attribute__((ext_vector_type(8)))  float         v8f;
typedef __attribute__((ext_vector_type(8)))  unsigned int  v8u;

// LDS strides (padded to break bank conflicts; keep 16B alignment)
#define SH_STRIDE   264   // 256 + 8 bf16
#define SQKV_STRIDE 776   // 768 + 8 bf16
#define SA_STRIDE   1032  // 1024 + 8 bf16

__device__ __forceinline__ float bf2f(unsigned short u) {
    unsigned int t = ((unsigned int)u) << 16;
    return __builtin_bit_cast(float, t);
}
__device__ __forceinline__ float bf2f_lo(unsigned int w) {
    unsigned int t = w << 16;
    return __builtin_bit_cast(float, t);
}
__device__ __forceinline__ float bf2f_hi(unsigned int w) {
    unsigned int t = w & 0xFFFF0000u;
    return __builtin_bit_cast(float, t);
}
__device__ __forceinline__ unsigned short f2bf(float f) {
    unsigned int x = __builtin_bit_cast(unsigned int, f);
    unsigned int r = x + 0x7FFFu + ((x >> 16) & 1u);   // round-to-nearest-even
    return (unsigned short)(r >> 16);
}
__device__ __forceinline__ unsigned int f2bf_pk(float lo, float hi) {
    return (unsigned int)f2bf(lo) | ((unsigned int)f2bf(hi) << 16);
}

// A fragment (16x32 bf16, M x K) from LDS, contiguous k order (k = 16*h + s), same as B:
// lane l: m = l&15, kbase = (l<16)?0:16 ; v0..7 = K[kb..kb+15] (contiguous 32B)
__device__ __forceinline__ v16bf load_fragA_lds(const unsigned short* s, int m0,
                                                int strideElems, int k0, int lane) {
    int row = m0 + (lane & 15);
    int kb  = (lane < 16) ? 0 : 16;
    const unsigned short* p = s + (size_t)row * strideElems + (k0 + kb);
    uint4 lo = *(const uint4*)(p);        // 8 bf16
    uint4 hi = *(const uint4*)(p + 8);    // 8 bf16, K+8
    v8u r;
    r[0] = lo.x; r[1] = lo.y; r[2] = lo.z; r[3] = lo.w;
    r[4] = hi.x; r[5] = hi.y; r[6] = hi.z; r[7] = hi.w;
    return __builtin_bit_cast(v16bf, r);
}

// B fragment (32x16 bf16, K x N) from bf16 weight stored row-major [N][K]:
// lane l: n = l&15, kbase = (l<16)?0:16 ; v0..7 = K[kb..kb+15] (contiguous 32B)
__device__ __forceinline__ v16bf load_fragB_w(const unsigned short* w, int n0,
                                              int Kdim, int k0, int lane) {
    int n  = n0 + (lane & 15);
    int kb = (lane < 16) ? 0 : 16;
    const unsigned short* p = w + (size_t)n * Kdim + (k0 + kb);
    uint4 lo = *(const uint4*)(p);
    uint4 hi = *(const uint4*)(p + 8);
    v8u r;
    r[0] = lo.x; r[1] = lo.y; r[2] = lo.z; r[3] = lo.w;
    r[4] = hi.x; r[5] = hi.y; r[6] = hi.z; r[7] = hi.w;
    return __builtin_bit_cast(v16bf, r);
}

__device__ __forceinline__ v8f wmma_bf16(v16bf a, v16bf b, v8f c) {
    return __builtin_amdgcn_wmma_f32_16x16x32_bf16(false, a, false, b, (short)0, c,
                                                   false, false);
}

// Unpack 32 bf16 (4 x uint4, 16B aligned) from LDS into fp32 registers
__device__ __forceinline__ void load32_bf16(const unsigned short* p, float* f) {
#pragma unroll
    for (int j = 0; j < 4; ++j) {
        uint4 u = *(const uint4*)(p + 8 * j);
        f[8 * j + 0] = bf2f_lo(u.x); f[8 * j + 1] = bf2f_hi(u.x);
        f[8 * j + 2] = bf2f_lo(u.y); f[8 * j + 3] = bf2f_hi(u.y);
        f[8 * j + 4] = bf2f_lo(u.z); f[8 * j + 5] = bf2f_hi(u.z);
        f[8 * j + 6] = bf2f_lo(u.w); f[8 * j + 7] = bf2f_hi(u.w);
    }
}

// ---------------------------------------------------------------------------
// Kernel A: fused LN1(+roll) -> QKV GEMM -> windowed attention -> proj GEMM
//           + shortcut (+inverse roll).  64 tokens / block, 256 threads.
// ---------------------------------------------------------------------------
__global__ __launch_bounds__(256) void swin_attn_kernel(
    const float* __restrict__ x, const float* __restrict__ xmask,
    const float* __restrict__ n1w, const float* __restrict__ n1b,
    const unsigned short* __restrict__ qkvw, const float* __restrict__ qkvb,
    const unsigned short* __restrict__ projw, const float* __restrict__ projb,
    float* __restrict__ y, int shift)
{
    extern __shared__ unsigned char smem_raw[];
    unsigned short* sH   = (unsigned short*)smem_raw;                        // 64 x 264
    unsigned short* sQKV = (unsigned short*)(smem_raw + 64 * SH_STRIDE * 2); // 64 x 776
    float*          sS   = (float*)(smem_raw + 64 * SH_STRIDE * 2 + 64 * SQKV_STRIDE * 2);

    const int g0   = blockIdx.x * 64;
    const int b    = g0 / L_;
    const int r0   = g0 % L_;          // rolled row base within batch
    const int wid  = threadIdx.x >> 5;
    const int lane = threadIdx.x & 31;
    const float NEG_INF = -__builtin_inff();

    // ---------------- Phase 1: LayerNorm (with roll gather) ----------------
    for (int rr = 0; rr < 8; ++rr) {
        int r    = wid * 8 + rr;                       // local row 0..63 (rolled)
        int orig = (r0 + r + shift) & (L_ - 1);
        const float* xr = x + ((size_t)b * L_ + orig) * C_;
        float xv[8];
        float sum = 0.f, sq = 0.f;
#pragma unroll
        for (int j = 0; j < 8; ++j) {
            float v = xr[lane + 32 * j];
            xv[j] = v; sum += v; sq += v * v;
        }
#pragma unroll
        for (int off = 16; off > 0; off >>= 1) {
            sum += __shfl_xor(sum, off, 32);
            sq  += __shfl_xor(sq,  off, 32);
        }
        float mean = sum * (1.0f / C_);
        float var  = sq * (1.0f / C_) - mean * mean;
        float rstd = rsqrtf(var + 1e-5f);
#pragma unroll
        for (int j = 0; j < 8; ++j) {
            int c = lane + 32 * j;
            float hv = (xv[j] - mean) * rstd * n1w[c] + n1b[c];
            sH[(size_t)r * SH_STRIDE + c] = f2bf(hv);
        }
    }
    if (threadIdx.x < 64) {
        int r    = threadIdx.x;
        int orig = (r0 + r + shift) & (L_ - 1);
        float m  = xmask[(size_t)b * L_ + orig];
        sS[r] = (shift > 0) ? (((orig < shift) ? 1.0f : 2.0f) * m) : m;
    }
    __syncthreads();

    // ---------------- Phase 2: QKV GEMM (M=64, N=768, K=256) ----------------
    // Wave owns one m-tile (2 waves per m-tile); preload all A frags once.
    {
        const int mt     = wid >> 1;                  // 0..3
        const int ntbase = (wid & 1) * 24;            // 24 n-tiles per wave
        v16bf afrag[8];
#pragma unroll
        for (int ks = 0; ks < 8; ++ks)
            afrag[ks] = load_fragA_lds(sH, mt * 16, SH_STRIDE, ks * 32, lane);
        for (int nt = ntbase; nt < ntbase + 24; ++nt) {
            v8f acc = {};
#pragma unroll
            for (int ks = 0; ks < 8; ++ks)
                acc = wmma_bf16(afrag[ks], load_fragB_w(qkvw, nt * 16, 256, ks * 32, lane), acc);
            int n    = nt * 16 + (lane & 15);
            float bi = qkvb[n];
            int mo   = mt * 16 + ((lane < 16) ? 0 : 8);
#pragma unroll
            for (int i = 0; i < 8; ++i)
                sQKV[(size_t)(mo + i) * SQKV_STRIDE + n] = f2bf(acc[i] + bi);
        }
    }
    __syncthreads();   // QKV ready; sH (h) dead -> reuse as attention output

    // ---------------- Phase 3: windowed attention (8x8 per head) -----------
    const float SCALE = 0.17677669529663687f;  // 32^-0.5
    for (int tt = 0; tt < 2; ++tt) {
        int task = threadIdx.x + tt * 256;     // 512 = 8 win * 8 head * 8 row
        int win  = task >> 6;
        int head = (task >> 3) & 7;
        int q    = task & 7;
        int rq   = win * 8 + q;

        float qv[32];
        load32_bf16(sQKV + (size_t)rq * SQKV_STRIDE + head * HD_, qv);

        float si = sS[rq];
        float sc[8];
        float mx = NEG_INF;
#pragma unroll
        for (int m = 0; m < 8; ++m) {
            int rk = win * 8 + m;
            float kv[32];
            load32_bf16(sQKV + (size_t)rk * SQKV_STRIDE + 256 + head * HD_, kv);
            float d0 = 0.f;
#pragma unroll
            for (int d = 0; d < 32; ++d) d0 += qv[d] * kv[d];
            float sj = sS[rk];
            bool masked = (shift > 0) ? (si != sj || si == 0.f || sj == 0.f)
                                      : (si * sj == 0.f);
            float s = masked ? NEG_INF : d0 * SCALE;
            sc[m] = s;
            mx = fmaxf(mx, s);
        }
        float out[32];
#pragma unroll
        for (int d = 0; d < 32; ++d) out[d] = 0.f;
        if (mx != NEG_INF) {                 // fully-masked row -> zeros (nan_to_num)
            float se = 0.f;
#pragma unroll
            for (int m = 0; m < 8; ++m) { float e = __expf(sc[m] - mx); sc[m] = e; se += e; }
            float inv = 1.0f / se;
#pragma unroll
            for (int m = 0; m < 8; ++m) {
                float p = sc[m] * inv;
                float vv[32];
                load32_bf16(sQKV + (size_t)(win * 8 + m) * SQKV_STRIDE + 512 + head * HD_, vv);
#pragma unroll
                for (int d = 0; d < 32; ++d) out[d] += p * vv[d];
            }
        }
        unsigned short* op = sH + (size_t)rq * SH_STRIDE + head * HD_;
#pragma unroll
        for (int j = 0; j < 4; ++j) {
            uint4 u;
            u.x = f2bf_pk(out[8 * j + 0], out[8 * j + 1]);
            u.y = f2bf_pk(out[8 * j + 2], out[8 * j + 3]);
            u.z = f2bf_pk(out[8 * j + 4], out[8 * j + 5]);
            u.w = f2bf_pk(out[8 * j + 6], out[8 * j + 7]);
            *(uint4*)(op + 8 * j) = u;
        }
    }
    __syncthreads();

    // ------------- Phase 4: proj GEMM + bias + shortcut (+roll back) -------
    {
        const int mt     = wid >> 1;                  // 0..3
        const int ntbase = (wid & 1) * 8;             // 8 n-tiles per wave
        v16bf afrag[8];
#pragma unroll
        for (int ks = 0; ks < 8; ++ks)
            afrag[ks] = load_fragA_lds(sH, mt * 16, SH_STRIDE, ks * 32, lane);
        for (int nt = ntbase; nt < ntbase + 8; ++nt) {
            v8f acc = {};
#pragma unroll
            for (int ks = 0; ks < 8; ++ks)
                acc = wmma_bf16(afrag[ks], load_fragB_w(projw, nt * 16, 256, ks * 32, lane), acc);
            int n    = nt * 16 + (lane & 15);
            float bi = projb[n];
            int mo   = mt * 16 + ((lane < 16) ? 0 : 8);
#pragma unroll
            for (int i = 0; i < 8; ++i) {
                int r    = mo + i;                                 // rolled row
                int orig = (r0 + r + shift) & (L_ - 1);            // roll(aw, +shift)
                size_t idx = ((size_t)b * L_ + orig) * C_ + n;
                y[idx] = acc[i] + bi + x[idx];                     // + shortcut
            }
        }
    }
}

// ---------------------------------------------------------------------------
// Kernel B: fused LN2 -> fc1 + GELU -> fc2 + bias + residual.
//           32 tokens / block, 256 threads.
// ---------------------------------------------------------------------------
__global__ __launch_bounds__(256) void swin_mlp_kernel(
    const float* __restrict__ y,
    const float* __restrict__ n2w, const float* __restrict__ n2b,
    const unsigned short* __restrict__ fc1w, const float* __restrict__ fc1b,
    const unsigned short* __restrict__ fc2w, const float* __restrict__ fc2b,
    float* __restrict__ xout)
{
    extern __shared__ unsigned char smem_raw[];
    unsigned short* sH = (unsigned short*)smem_raw;                        // 32 x 264
    unsigned short* sA = (unsigned short*)(smem_raw + 32 * SH_STRIDE * 2); // 32 x 1032

    const int g0   = blockIdx.x * 32;
    const int wid  = threadIdx.x >> 5;
    const int lane = threadIdx.x & 31;

    // ---------------- Phase 1: LayerNorm2 ----------------------------------
    for (int rr = 0; rr < 4; ++rr) {
        int r = wid * 4 + rr;
        const float* yr = y + (size_t)(g0 + r) * C_;
        float xv[8];
        float sum = 0.f, sq = 0.f;
#pragma unroll
        for (int j = 0; j < 8; ++j) {
            float v = yr[lane + 32 * j];
            xv[j] = v; sum += v; sq += v * v;
        }
#pragma unroll
        for (int off = 16; off > 0; off >>= 1) {
            sum += __shfl_xor(sum, off, 32);
            sq  += __shfl_xor(sq,  off, 32);
        }
        float mean = sum * (1.0f / C_);
        float var  = sq * (1.0f / C_) - mean * mean;
        float rstd = rsqrtf(var + 1e-5f);
#pragma unroll
        for (int j = 0; j < 8; ++j) {
            int c = lane + 32 * j;
            float hv = (xv[j] - mean) * rstd * n2w[c] + n2b[c];
            sH[(size_t)r * SH_STRIDE + c] = f2bf(hv);
        }
    }
    __syncthreads();

    // ---------------- Phase 2: fc1 (M=32, N=1024, K=256) + GELU ------------
    // 4 waves per m-tile; preload A frags once, sweep 16 n-tiles.
    {
        const int mt     = wid >> 2;                  // 0..1
        const int ntbase = (wid & 3) * 16;            // 16 n-tiles per wave
        v16bf afrag[8];
#pragma unroll
        for (int ks = 0; ks < 8; ++ks)
            afrag[ks] = load_fragA_lds(sH, mt * 16, SH_STRIDE, ks * 32, lane);
        for (int nt = ntbase; nt < ntbase + 16; ++nt) {
            v8f acc = {};
#pragma unroll
            for (int ks = 0; ks < 8; ++ks)
                acc = wmma_bf16(afrag[ks], load_fragB_w(fc1w, nt * 16, 256, ks * 32, lane), acc);
            int n    = nt * 16 + (lane & 15);
            float bi = fc1b[n];
            int mo   = mt * 16 + ((lane < 16) ? 0 : 8);
#pragma unroll
            for (int i = 0; i < 8; ++i) {
                float g  = acc[i] + bi;
                float gl = 0.5f * g * (1.0f + erff(g * 0.70710678118654752f)); // exact GELU
                sA[(size_t)(mo + i) * SA_STRIDE + n] = f2bf(gl);
            }
        }
    }
    __syncthreads();

    // ---------------- Phase 3: fc2 (M=32, N=256, K=1024) + residual --------
    // K chunked 4 x 8 frags; 4 n-tiles per wave with accumulators held
    // across chunks so each A fragment is reused 4x.
    {
        const int mt     = wid >> 2;                  // 0..1
        const int ntbase = (wid & 3) * 4;             // 4 n-tiles per wave
        v8f acc[4] = {{}, {}, {}, {}};
        for (int kc = 0; kc < 4; ++kc) {
            v16bf afrag[8];
#pragma unroll
            for (int j = 0; j < 8; ++j)
                afrag[j] = load_fragA_lds(sA, mt * 16, SA_STRIDE, (kc * 8 + j) * 32, lane);
#pragma unroll
            for (int ntl = 0; ntl < 4; ++ntl) {
                int nt = ntbase + ntl;
#pragma unroll
                for (int j = 0; j < 8; ++j)
                    acc[ntl] = wmma_bf16(afrag[j],
                                         load_fragB_w(fc2w, nt * 16, 1024, (kc * 8 + j) * 32, lane),
                                         acc[ntl]);
            }
        }
#pragma unroll
        for (int ntl = 0; ntl < 4; ++ntl) {
            int n    = (ntbase + ntl) * 16 + (lane & 15);
            float bi = fc2b[n];
            int mo   = mt * 16 + ((lane < 16) ? 0 : 8);
#pragma unroll
            for (int i = 0; i < 8; ++i) {
                size_t idx = (size_t)(g0 + mo + i) * C_ + n;
                xout[idx] = acc[ntl][i] + bi + y[idx];
            }
        }
    }
}

// ---------------------------------------------------------------------------
__global__ void cast_f32_bf16(const float* __restrict__ src,
                              unsigned short* __restrict__ dst, int n)
{
    int i = blockIdx.x * 256 + threadIdx.x;
    if (i < n) dst[i] = f2bf(src[i]);
}

// ---------------------------------------------------------------------------
extern "C" void kernel_launch(void* const* d_in, const int* in_sizes, int n_in,
                              void* d_out, int out_size, void* d_ws, size_t ws_size,
                              hipStream_t stream)
{
    (void)in_sizes; (void)n_in; (void)out_size; (void)ws_size;

    const float* x     = (const float*)d_in[0];
    const float* xmask = (const float*)d_in[1];
    const float* n1w   = (const float*)d_in[2];
    const float* n1b   = (const float*)d_in[3];
    const float* n2w   = (const float*)d_in[4];
    const float* n2b   = (const float*)d_in[5];
    const float* qkvwf = (const float*)d_in[6];
    const float* qkvb  = (const float*)d_in[7];
    const float* projwf= (const float*)d_in[8];
    const float* projb = (const float*)d_in[9];
    const float* fc1wf = (const float*)d_in[10];
    const float* fc1b  = (const float*)d_in[11];
    const float* fc2wf = (const float*)d_in[12];
    const float* fc2b  = (const float*)d_in[13];
    float* out = (float*)d_out;

    unsigned char* ws = (unsigned char*)d_ws;
    float* y = (float*)ws;
    size_t off = (size_t)B_ * L_ * C_ * sizeof(float);
    unsigned short* qkvw = (unsigned short*)(ws + off); off += (size_t)768 * 256 * 2;
    unsigned short* projw = (unsigned short*)(ws + off); off += (size_t)256 * 256 * 2;
    unsigned short* fc1w = (unsigned short*)(ws + off); off += (size_t)1024 * 256 * 2;
    unsigned short* fc2w = (unsigned short*)(ws + off);

    // Cast weights to bf16 once per launch (deterministic; ~1.5 MB total)
    cast_f32_bf16<<<(768 * 256 + 255) / 256, 256, 0, stream>>>(qkvwf, qkvw, 768 * 256);
    cast_f32_bf16<<<(256 * 256 + 255) / 256, 256, 0, stream>>>(projwf, projw, 256 * 256);
    cast_f32_bf16<<<(1024 * 256 + 255) / 256, 256, 0, stream>>>(fc1wf, fc1w, 1024 * 256);
    cast_f32_bf16<<<(256 * 1024 + 255) / 256, 256, 0, stream>>>(fc2wf, fc2w, 256 * 1024);

    const size_t smemA = (size_t)64 * SH_STRIDE * 2 + (size_t)64 * SQKV_STRIDE * 2 + 64 * 4;
    const size_t smemB = (size_t)32 * SH_STRIDE * 2 + (size_t)32 * SA_STRIDE * 2;
    const int gridA = B_ * L_ / 64;   // 2048
    const int gridB = B_ * L_ / 32;   // 4096

    // Pass 1: plain windows (shift = 0). x -> y -> out (= x1)
    swin_attn_kernel<<<gridA, 256, smemA, stream>>>(x, xmask, n1w, n1b,
                                                    qkvw, qkvb, projw, projb, y, 0);
    swin_mlp_kernel<<<gridB, 256, smemB, stream>>>(y, n2w, n2b, fc1w, fc1b,
                                                   fc2w, fc2b, out);

    // Pass 2: shifted windows (shift = 4). out (= x1) -> y -> out (final)
    swin_attn_kernel<<<gridA, 256, smemA, stream>>>(out, xmask, n1w, n1b,
                                                    qkvw, qkvb, projw, projb, y, SHIFT_);
    swin_mlp_kernel<<<gridB, 256, smemB, stream>>>(y, n2w, n2b, fc1w, fc1b,
                                                   fc2w, fc2b, out);
}
